// SPGCrossAttention_38044820308469
// MI455X (gfx1250) — compile-verified, matrix operand order checked by reading, not run on hardware
//
#include <hip/hip_runtime.h>
#include <hip/hip_bf16.h>

// ---------------------------------------------------------------------------
// SPG cross-attention for MI455X (gfx1250), wave32 + WMMA f16->f32.
//   out_A = WoutA @ attn(qB, kA, vA) + boutA + GN(x_A)
//   out_B = WoutB @ attn(qA, kB, vB) + boutB + GN(x_B)
// All WMMA fragments are fed by 16-byte vector loads (global_load_b128 /
// ds_load_b128) via transposed producer layouts:
//   norm f16   : (B, S, C)   spatial-major  -> GEMM B-frags are contiguous
//   Q, K       : (B,H,S,32)  row-major      -> score frags contiguous
//   V          : (B,H,32,S)  d-major        -> P@V B-frags contiguous
//   attn f16   : (B, S, C)   spatial-major  -> out-GEMM B-frags contiguous
// GEMM / attention inner loops are software-pipelined (next-step fragments
// prefetched into separate registers) so global loads overlap WMMA issue.
// ---------------------------------------------------------------------------

#define BATCH 16
#define CH    256
#define SP    1024     // H*W = 32*32
#define NHEAD 8
#define HDIM  32
#define GRP   16
#define EPSF  1e-5f

typedef __attribute__((ext_vector_type(16))) _Float16 v16h;
typedef __attribute__((ext_vector_type(8)))  _Float16 v8h;
typedef __attribute__((ext_vector_type(8)))  float    v8f;

// ---------------------------------------------------------------------------
// WMMA fragment helpers (CDNA5 ISA 7.12.2 16-bit layouts, wave32).
//   A (16xK=32):  row = lane&15 ; lanes 0-15 hold K {0..7,16..23},
//                                 lanes 16-31 hold K {8..15,24..31}
//   B (K=32x16):  col = lane&15 ; same K split across lane halves
//   C/D f32:      VGPR r -> row r (lanes 0-15) / row r+8 (lanes 16-31),
//                 col = lane&15
// Fragment slot j -> k = (j<8 ? j : j+8) + (lane>=16 ? 8 : 0)
// ---------------------------------------------------------------------------
__device__ __forceinline__ v16h pack_frag(v8h lo, v8h hi) {
  v16h r;
#pragma unroll
  for (int j = 0; j < 8; ++j) { r[j] = lo[j]; r[j + 8] = hi[j]; }
  return r;
}

// Fragment from row-major tile: element(row, k) = p[row*ld + k]; row=lane&15.
// Works for A (row=M) and for B when the producer stored B transposed (row=N).
// Requires ld % 8 == 0 and 16B-aligned tile base.
__device__ __forceinline__ v16h load_frag_mk(const _Float16* __restrict__ p, int ld) {
  const int lane = threadIdx.x & 31;
  const int row  = lane & 15;
  const int koff = (lane >> 4) << 3;
  const _Float16* q = p + (size_t)row * ld + koff;
  v8h lo = *reinterpret_cast<const v8h*>(q);
  v8h hi = *reinterpret_cast<const v8h*>(q + 16);
  return pack_frag(lo, hi);
}

__device__ __forceinline__ v8f wmma_f32(v16h a, v16h b, v8f c) {
  return __builtin_amdgcn_wmma_f32_16x16x32_f16(
      /*neg_a=*/false, a, /*neg_b=*/false, b,
      /*c_mod=*/(short)0, c, /*reuse_a=*/false, /*reuse_b=*/false);
}

// ---------------------------------------------------------------------------
// f32 -> f16 weight conversion
// ---------------------------------------------------------------------------
__global__ void cvt_f16_kernel(const float* __restrict__ src,
                               _Float16* __restrict__ dst, int n) {
  int i = blockIdx.x * 256 + threadIdx.x;
  if (i < n) dst[i] = (_Float16)src[i];
}

// ---------------------------------------------------------------------------
// GroupNorm: one block per (batch, group); 16 ch x 1024 spatial per group.
// Emits fp32 (B,C,S) for the residual path and f16 **transposed** (B,S,C)
// for the GEMM path.
// ---------------------------------------------------------------------------
__global__ void __launch_bounds__(256)
groupnorm_kernel(const float* __restrict__ x,
                 const float* __restrict__ gamma, const float* __restrict__ beta,
                 float* __restrict__ y32,          // (B, C, S)
                 _Float16* __restrict__ y16T) {    // (B, S, C)
  const int g = blockIdx.x % GRP;
  const int b = blockIdx.x / GRP;
  const int cpg = CH / GRP;              // 16
  const int n   = cpg * SP;              // 16384
  const size_t base = ((size_t)b * CH + (size_t)g * cpg) * SP;
  const float* xg = x + base;

  __shared__ float red0[256];
  __shared__ float red1[256];
  float s0 = 0.f, s1 = 0.f;
  for (int i = threadIdx.x; i < n; i += 256) {
    float v = xg[i];
    s0 += v; s1 += v * v;
  }
  red0[threadIdx.x] = s0; red1[threadIdx.x] = s1;
  __syncthreads();
  for (int off = 128; off > 0; off >>= 1) {
    if ((int)threadIdx.x < off) {
      red0[threadIdx.x] += red0[threadIdx.x + off];
      red1[threadIdx.x] += red1[threadIdx.x + off];
    }
    __syncthreads();
  }
  const float mu   = red0[0] / (float)n;
  const float var  = red1[0] / (float)n - mu * mu;
  const float rstd = rsqrtf(var + EPSF);

  float*    y32g = y32 + base;
  _Float16* yTb  = y16T + (size_t)b * SP * CH;
  for (int i = threadIdx.x; i < n; i += 256) {
    int ci = i / SP, s = i % SP;
    int c = g * cpg + ci;
    float v = (xg[i] - mu) * rstd * gamma[c] + beta[c];
    y32g[i] = v;
    yTb[(size_t)s * CH + c] = (_Float16)v;
  }
}

// ---------------------------------------------------------------------------
// Software-pipelined 16x64-strip GEMM core: acc[0..3] += W_tile @ XT_strip.
// One A-frag shared by 4 B-frags per K-step; next step's 5 fragments are
// prefetched while the current 4 WMMAs issue.
// ---------------------------------------------------------------------------
__device__ __forceinline__ void gemm_strip(const _Float16* __restrict__ Wt,
                                           const _Float16* __restrict__ xt0,
                                           v8f& acc0, v8f& acc1,
                                           v8f& acc2, v8f& acc3) {
  v16h a  = load_frag_mk(Wt, CH);
  v16h b0 = load_frag_mk(xt0,           CH);
  v16h b1 = load_frag_mk(xt0 + 16 * CH, CH);
  v16h b2 = load_frag_mk(xt0 + 32 * CH, CH);
  v16h b3 = load_frag_mk(xt0 + 48 * CH, CH);
#pragma unroll
  for (int kc = 0; kc < CH / 32; ++kc) {
    v16h an = a, bn0 = b0, bn1 = b1, bn2 = b2, bn3 = b3;
    if (kc + 1 < CH / 32) {
      const _Float16* xn = xt0 + (kc + 1) * 32;
      an  = load_frag_mk(Wt + (kc + 1) * 32, CH);
      bn0 = load_frag_mk(xn,           CH);
      bn1 = load_frag_mk(xn + 16 * CH, CH);
      bn2 = load_frag_mk(xn + 32 * CH, CH);
      bn3 = load_frag_mk(xn + 48 * CH, CH);
    }
    acc0 = wmma_f32(a, b0, acc0);
    acc1 = wmma_f32(a, b1, acc1);
    acc2 = wmma_f32(a, b2, acc2);
    acc3 = wmma_f32(a, b3, acc3);
    a = an; b0 = bn0; b1 = bn1; b2 = bn2; b3 = bn3;
  }
}

// ---------------------------------------------------------------------------
// QKV projection: per batch  Y(768,1024) = Wqkv(768,256) @ norm(256,1024).
// 8 waves/block; each wave computes a 16x64 strip.
// Stores: q,k -> (B,NH,{q,k},S,32) with packed v8h stores; v -> (B,NH,32,S).
// ---------------------------------------------------------------------------
__global__ void __launch_bounds__(256)
qkv_gemm_kernel(const _Float16* __restrict__ W,      // (3C, C) f16
                const _Float16* __restrict__ XT,     // (B, S, C) f16
                _Float16* __restrict__ qkv) {        // (B, NH, 3, ...) f16
  const int MT = (3 * CH) / 16;                      // 48
  const int NG = SP / 64;                            // 16 strips of 64 cols
  int gid = blockIdx.x * 8 + (threadIdx.x >> 5);
  int b  = gid / (MT * NG);
  int r  = gid % (MT * NG);
  int mt = r / NG, ng = r % NG;
  const _Float16* XTb = XT + (size_t)b * SP * CH;

  v8f acc0 = {}, acc1 = {}, acc2 = {}, acc3 = {};
  gemm_strip(W + (size_t)(mt * 16) * CH,
             XTb + (size_t)(ng * 64) * CH, acc0, acc1, acc2, acc3);

  const int lane = threadIdx.x & 31;
  const int n0 = lane & 15, hh = lane >> 4;
  // tile row block: channels [mt*16, mt*16+16) of 768; type fixed per tile
  const int cRow0 = mt * 16;
  const int head  = cRow0 / 96;
  const int w0    = cRow0 % 96;
  const int t     = w0 / 32;              // 0=q, 1=k, 2=v
  const int chB   = (w0 % 32) + hh * 8;   // this lane's first channel-in-head
  const size_t hb = ((size_t)b * NHEAD + head) * 3;

  v8f accs[4] = {acc0, acc1, acc2, acc3};
#pragma unroll
  for (int ti = 0; ti < 4; ++ti) {
    int sCol = ng * 64 + ti * 16 + n0;
    if (t < 2) {
      // (S, 32) layout: lane's 8 values are 8 consecutive channels -> v8h store
      v8h pk;
#pragma unroll
      for (int rr = 0; rr < 8; ++rr) pk[rr] = (_Float16)accs[ti][rr];
      _Float16* dst = qkv + ((hb + t) * SP + sCol) * HDIM + chB;
      *reinterpret_cast<v8h*>(dst) = pk;
    } else {
      // V: (32, S) layout for contiguous P@V B-fragments
      _Float16* dst = qkv + (hb + 2) * SP * HDIM;
#pragma unroll
      for (int rr = 0; rr < 8; ++rr)
        dst[(size_t)(chB + rr) * SP + sCol] = (_Float16)accs[ti][rr];
    }
  }
}

// ---------------------------------------------------------------------------
// Attention: one wave per (b, head, 16-query block).
//   pass1: scores(16,1024) = Qblk(16,32) @ K^T -> f16 strip in LDS (32KB)
//   softmax on v8h chunks (two lanes/row, shfl_xor(16) combine)
//   pass2: O(16,32) = P(16,1024) @ V(32,1024)^T ; A-frags = 2x ds_load_b128
// d=32 == WMMA K exactly -> one WMMA per 16x16 score tile.
// Both WMMA loops prefetch the next tile's fragments (pipelined).
// ---------------------------------------------------------------------------
__global__ void __launch_bounds__(32)
attn_kernel(const _Float16* __restrict__ qkvQ,   // branch supplying Q
            const _Float16* __restrict__ qkvKV,  // branch supplying K, V
            _Float16* __restrict__ attnT) {      // (B, S, C) f16
  extern __shared__ _Float16 pm[];               // 16 * 1024 f16 = 32KB

  int id = blockIdx.x;
  const int QB = SP / 16;                        // 64
  int qb = id % QB; id /= QB;
  int h  = id % NHEAD;
  int b  = id / NHEAD;

  const size_t hb = ((size_t)b * NHEAD + h) * 3;
  const _Float16* Q = qkvQ  + (hb + 0) * SP * HDIM;   // (S, 32)
  const _Float16* K = qkvKV + (hb + 1) * SP * HDIM;   // (S, 32)
  const _Float16* V = qkvKV + (hb + 2) * SP * HDIM;   // (32, S)

  const int lane = threadIdx.x & 31;
  const int n0 = lane & 15, hh = lane >> 4;
  const int koff = hh << 3;

  // ---- pass 1: scores -> f16 LDS strip (pipelined K-tile loads) ----
  v16h aq = load_frag_mk(Q + (size_t)qb * 16 * HDIM, HDIM);
  {
    v16h bk = load_frag_mk(K, HDIM);
    for (int kb = 0; kb < SP / 16; ++kb) {
      v16h bn = bk;
      if (kb + 1 < SP / 16)
        bn = load_frag_mk(K + (size_t)(kb + 1) * 16 * HDIM, HDIM);
      v8f c = {};
      c = wmma_f32(aq, bk, c);
#pragma unroll
      for (int rr = 0; rr < 8; ++rr)
        pm[(size_t)(rr + hh * 8) * SP + kb * 16 + n0] =
            (_Float16)(c[rr] * 0.0625f);           // 1/sqrt(256)
      bk = bn;
    }
  }
  __syncthreads();

  // ---- softmax: lane pair (L, L^16) shares row n0, halves of 512 cols ----
  {
    v8h* rowp = reinterpret_cast<v8h*>(pm + (size_t)n0 * SP + hh * 512);
    float mx = -3.402823466e+38f;
    for (int i = 0; i < 64; ++i) {
      v8h v = rowp[i];
#pragma unroll
      for (int j = 0; j < 8; ++j) mx = fmaxf(mx, (float)v[j]);
    }
    mx = fmaxf(mx, __shfl_xor(mx, 16, 32));
    float sum = 0.f;
    for (int i = 0; i < 64; ++i) {
      v8h v = rowp[i];
      v8h e;
#pragma unroll
      for (int j = 0; j < 8; ++j) {
        float ef = __expf((float)v[j] - mx);
        e[j] = (_Float16)ef;
        sum += ef;
      }
      rowp[i] = e;
    }
    sum += __shfl_xor(sum, 16, 32);
    float inv = 1.0f / sum;
    for (int i = 0; i < 64; ++i) {
      v8h v = rowp[i];
#pragma unroll
      for (int j = 0; j < 8; ++j) v[j] = (_Float16)((float)v[j] * inv);
      rowp[i] = v;
    }
  }
  __syncthreads();

  // ---- pass 2: O = P @ V (pipelined V-tile + LDS-P loads) ----
  v8f o0 = {}, o1 = {};
  {
    const _Float16* pr0 = pm + (size_t)n0 * SP + koff;
    v8h plo = *reinterpret_cast<const v8h*>(pr0);
    v8h phi = *reinterpret_cast<const v8h*>(pr0 + 16);
    v16h ap = pack_frag(plo, phi);
    v16h b0 = load_frag_mk(V, SP);
    v16h b1 = load_frag_mk(V + (size_t)16 * SP, SP);
    for (int kc = 0; kc < SP / 32; ++kc) {
      v16h apn = ap, bn0 = b0, bn1 = b1;
      if (kc + 1 < SP / 32) {
        const _Float16* pr = pm + (size_t)n0 * SP + (kc + 1) * 32 + koff;
        v8h nlo = *reinterpret_cast<const v8h*>(pr);
        v8h nhi = *reinterpret_cast<const v8h*>(pr + 16);
        apn = pack_frag(nlo, nhi);
        bn0 = load_frag_mk(V + (kc + 1) * 32,                    SP);
        bn1 = load_frag_mk(V + (size_t)16 * SP + (kc + 1) * 32,  SP);
      }
      o0 = wmma_f32(ap, b0, o0);
      o1 = wmma_f32(ap, b1, o1);
      ap = apn; b0 = bn0; b1 = bn1;
    }
  }

  // ---- store transposed: attnT[b][spatial][h*32 + ch] ----
  _Float16* outB = attnT + (size_t)b * SP * CH;
#pragma unroll
  for (int rr = 0; rr < 8; ++rr) {
    int sp = qb * 16 + rr + hh * 8;
    outB[(size_t)sp * CH + h * HDIM + n0]      = (_Float16)o0[rr];
    outB[(size_t)sp * CH + h * HDIM + 16 + n0] = (_Float16)o1[rr];
  }
}

// ---------------------------------------------------------------------------
// Output projection + bias + GroupNorm residual:
//   out(b,c,s) = sum_k Wout(c,k) * attn(b,k,s) + bias(c) + norm32(b,c,s)
// attn consumed spatial-major -> contiguous B-fragments.
// ---------------------------------------------------------------------------
__global__ void __launch_bounds__(256)
out_gemm_kernel(const _Float16* __restrict__ W,       // (C, C) f16
                const _Float16* __restrict__ XattnT,  // (B, S, C) f16
                const float* __restrict__ bias,       // (C)
                const float* __restrict__ resid,      // (B, C, S) f32
                float* __restrict__ out) {            // (B, C, S) f32
  const int MT = CH / 16;                             // 16
  const int NG = SP / 64;                             // 16
  int gid = blockIdx.x * 8 + (threadIdx.x >> 5);
  int b  = gid / (MT * NG);
  int r  = gid % (MT * NG);
  int mt = r / NG, ng = r % NG;
  const _Float16* XTb = XattnT + (size_t)b * SP * CH;

  v8f acc0 = {}, acc1 = {}, acc2 = {}, acc3 = {};
  gemm_strip(W + (size_t)(mt * 16) * CH,
             XTb + (size_t)(ng * 64) * CH, acc0, acc1, acc2, acc3);

  const int lane = threadIdx.x & 31;
  const int n0 = lane & 15, hh = lane >> 4;
  v8f accs[4] = {acc0, acc1, acc2, acc3};
#pragma unroll
  for (int ti = 0; ti < 4; ++ti) {
    int sCol = ng * 64 + ti * 16 + n0;
#pragma unroll
    for (int rr = 0; rr < 8; ++rr) {
      int cRow = mt * 16 + rr + hh * 8;
      size_t idx = ((size_t)b * CH + cRow) * SP + sCol;
      out[idx] = accs[ti][rr] + bias[cRow] + resid[idx];
    }
  }
}

// ---------------------------------------------------------------------------
// Host-side orchestration
// ---------------------------------------------------------------------------
extern "C" void kernel_launch(void* const* d_in, const int* in_sizes, int n_in,
                              void* d_out, int out_size, void* d_ws, size_t ws_size,
                              hipStream_t stream) {
  const float* xA    = (const float*)d_in[0];
  const float* xB    = (const float*)d_in[1];
  const float* gA    = (const float*)d_in[2];
  const float* bA    = (const float*)d_in[3];
  const float* gB    = (const float*)d_in[4];
  const float* bB    = (const float*)d_in[5];
  const float* WqkvA = (const float*)d_in[6];
  const float* WqkvB = (const float*)d_in[7];
  const float* WoutA = (const float*)d_in[8];
  const float* boutA = (const float*)d_in[9];
  const float* WoutB = (const float*)d_in[10];
  const float* boutB = (const float*)d_in[11];
  float* out = (float*)d_out;

  const size_t BCS = (size_t)BATCH * CH * SP;

  // workspace carve-out (256B aligned)
  char* w = (char*)d_ws;
  size_t off = 0;
  auto alloc = [&](size_t bytes) -> void* {
    void* p = w + off;
    off = (off + bytes + 255) & ~(size_t)255;
    return p;
  };
  float*    normA32 = (float*)alloc(BCS * 4);          // (B,C,S)
  float*    normB32 = (float*)alloc(BCS * 4);
  _Float16* normA16 = (_Float16*)alloc(BCS * 2);       // (B,S,C)
  _Float16* normB16 = (_Float16*)alloc(BCS * 2);
  _Float16* qkvA    = (_Float16*)alloc(3 * BCS * 2);   // (B,NH,3,...)
  _Float16* qkvB    = (_Float16*)alloc(3 * BCS * 2);
  _Float16* attnA   = (_Float16*)alloc(BCS * 2);       // (B,S,C)
  _Float16* attnB   = (_Float16*)alloc(BCS * 2);
  _Float16* WqkvA16 = (_Float16*)alloc((size_t)3 * CH * CH * 2);
  _Float16* WqkvB16 = (_Float16*)alloc((size_t)3 * CH * CH * 2);
  _Float16* WoutA16 = (_Float16*)alloc((size_t)CH * CH * 2);
  _Float16* WoutB16 = (_Float16*)alloc((size_t)CH * CH * 2);
  (void)ws_size; (void)in_sizes; (void)n_in; (void)out_size;

  // 1) weights -> f16
  {
    int nqkv = 3 * CH * CH, nout = CH * CH;
    cvt_f16_kernel<<<(nqkv + 255) / 256, 256, 0, stream>>>(WqkvA, WqkvA16, nqkv);
    cvt_f16_kernel<<<(nqkv + 255) / 256, 256, 0, stream>>>(WqkvB, WqkvB16, nqkv);
    cvt_f16_kernel<<<(nout + 255) / 256, 256, 0, stream>>>(WoutA, WoutA16, nout);
    cvt_f16_kernel<<<(nout + 255) / 256, 256, 0, stream>>>(WoutB, WoutB16, nout);
  }

  // 2) GroupNorm both branches
  groupnorm_kernel<<<BATCH * GRP, 256, 0, stream>>>(xA, gA, bA, normA32, normA16);
  groupnorm_kernel<<<BATCH * GRP, 256, 0, stream>>>(xB, gB, bB, normB32, normB16);

  // 3) QKV projections: 48 Mtiles x 16 Ngroups x 16 batches, 8 waves/block
  {
    int blocks = (48 * 16 * BATCH) / 8;  // 1536
    qkv_gemm_kernel<<<blocks, 256, 0, stream>>>(WqkvA16, normA16, qkvA);
    qkv_gemm_kernel<<<blocks, 256, 0, stream>>>(WqkvB16, normB16, qkvB);
  }

  // 4) cross attention: A-branch uses qB with kA/vA; B-branch uses qA with kB/vB
  {
    int blocks = BATCH * NHEAD * (SP / 16);           // 8192, 1 wave each
    size_t lds = (size_t)16 * SP * sizeof(_Float16); // 32KB f16 strip
    attn_kernel<<<blocks, 32, lds, stream>>>(qkvB, qkvA, attnA);
    attn_kernel<<<blocks, 32, lds, stream>>>(qkvA, qkvB, attnB);
  }

  // 5) output projection + bias + residual
  {
    int blocks = (16 * 16 * BATCH) / 8;  // 512
    out_gemm_kernel<<<blocks, 256, 0, stream>>>(WoutA16, attnA, boutA, normA32, out);
    out_gemm_kernel<<<blocks, 256, 0, stream>>>(WoutB16, attnB, boutB, normB32, out + BCS);
  }
}
